// OPA_13091060318578
// MI455X (gfx1250) — matrix operand order repaired and checked by reading, not compiled, not run on hardware
//
#include <hip/hip_runtime.h>
#include <hip/hip_bf16.h>
#include <math.h>

typedef __attribute__((ext_vector_type(16))) __bf16 v16bf;
typedef __attribute__((ext_vector_type(8)))  float  v8f;
typedef __attribute__((ext_vector_type(4)))  unsigned uv4;
typedef __attribute__((ext_vector_type(8)))  int      iv8;
typedef __attribute__((ext_vector_type(4)))  int      iv4;

#define TQ 2048
#define TK 2048
#define DDIM 512

// ---------- bf16 helpers (RNE), storage-only: no __bf16 arithmetic ----------
__device__ inline unsigned short f2bf(float f) {
  union { float f; unsigned u; } x; x.f = f;
  unsigned u = x.u + 0x7fffu + ((x.u >> 16) & 1u);
  return (unsigned short)(u >> 16);
}
__device__ inline float bf2f(unsigned short h) {
  union { unsigned u; float f; } x; x.u = ((unsigned)h) << 16;
  return x.f;
}

union AFrag { uint4 q[2]; unsigned u[8]; v16bf v; };

// ---------------------------------------------------------------------------
// TDM: bulk-load a 16-row x 512-element bf16 tile (contiguous rows, row
// stride 512) from global memory into LDS. Descriptor layout per CDNA5 ISA
// "Tensor DMA Descriptor (D#)" groups 0/1 (2D tensor, remaining groups zero).
// 6-arg builtin form (clang-23 / therock-10.0 headers).
// ---------------------------------------------------------------------------
__device__ inline void tdm_load_16x512_bf16(unsigned lds_off, const void* g) {
  unsigned long long ga = (unsigned long long)g;
  uv4 g0;
  g0[0] = 1u;                                            // count=1, no gather
  g0[1] = lds_off;                                       // lds_addr
  g0[2] = (unsigned)(ga & 0xffffffffu);                  // global_addr[31:0]
  g0[3] = (unsigned)((ga >> 32) & 0x01ffffffu)           // global_addr[56:32]
        | (2u << 30);                                    // type=2 ("image")
  iv8 g1;
  g1[0] = (int)(1u << 16);        // wg_mask=0, data_size=1 (2 bytes)
  g1[1] = (int)(512u << 16);      // tensor_dim0[15:0]=512 (bits 63:48)
  g1[2] = (int)(2048u << 16);     // dim0 hi=0 | tensor_dim1[15:0]=2048
  g1[3] = (int)(512u << 16);      // dim1 hi=0 | tile_dim0=512
  g1[4] = 16;                     // tile_dim1=16, tile_dim2=0
  g1[5] = 512;                    // tensor_dim0_stride lo32 = 512 elements
  g1[6] = 0;                      // stride hi | tensor_dim1_stride lo
  g1[7] = 0;
  iv4 g2 = {0, 0, 0, 0}, g3 = {0, 0, 0, 0};
  iv8 g4 = {0, 0, 0, 0, 0, 0, 0, 0};
  __builtin_amdgcn_tensor_load_to_lds(g0, g1, g2, g3, g4, 0);
}

// ============================================================================
// Kernel 0: convert seq_q/seq_k -> bf16, compute fp32 row sum-of-squares,
//           zero the global accumulators.
// ============================================================================
__global__ __launch_bounds__(256)
void prep_kernel(const float* __restrict__ seq_q, const float* __restrict__ seq_k,
                 unsigned short* __restrict__ qb, unsigned short* __restrict__ kb,
                 float* __restrict__ qn, float* __restrict__ kn,
                 float* __restrict__ acc) {
  int b = blockIdx.x;
  if (b == TQ + TK) { if (threadIdx.x < 2) acc[threadIdx.x] = 0.0f; return; }
  const float* src; unsigned short* dst; float* nrm; int row;
  if (b < TQ) { src = seq_q; dst = qb; nrm = qn; row = b; }
  else        { src = seq_k; dst = kb; nrm = kn; row = b - TQ; }
  __shared__ float red[256];
  float ss = 0.0f;
  for (int i = threadIdx.x; i < DDIM; i += 256) {
    float f = src[row * DDIM + i];
    dst[row * DDIM + i] = f2bf(f);
    ss += f * f;
  }
  red[threadIdx.x] = ss; __syncthreads();
  for (int s = 128; s > 0; s >>= 1) {
    if (threadIdx.x < s) red[threadIdx.x] += red[threadIdx.x + s];
    __syncthreads();
  }
  if (threadIdx.x == 0) nrm[row] = red[0];
}

// ============================================================================
// Kernel 1: D = sqrt(max(|q|^2 + |k|^2 - 2 q.k, 0)) via bf16 WMMA GEMM.
// Block = 8 waves covering one 16-row M stripe x 128 columns.
// TDM stages the shared A tile (wave 0) and each wave's B tile into LDS;
// fragment loads are ds_load_b128, K swept 32 at a time.
// ============================================================================
__global__ __launch_bounds__(256)
void dist_kernel(const unsigned short* __restrict__ qb,
                 const unsigned short* __restrict__ kb,
                 const float* __restrict__ qn, const float* __restrict__ kn,
                 float* __restrict__ Dmat) {
  __shared__ alignas(16) unsigned short sQ[16 * DDIM];       // 16 KB (A tile)
  __shared__ alignas(16) unsigned short sB[8][16 * DDIM];    // 128 KB (B tiles)

  const int wv   = threadIdx.x >> 5;
  const int lane = threadIdx.x & 31;
  const int kh   = lane >> 4;      // lane half
  const int l15  = lane & 15;
  const int tile = blockIdx.x * 8 + wv;
  const int m0 = (tile >> 7) << 4;     // same for all 8 waves in the block
  const int n0 = (tile & 127) << 4;

  // Async tensor DMA into LDS, then wait + barrier.
  if (wv == 0)
    tdm_load_16x512_bf16((unsigned)(size_t)&sQ[0], qb + (size_t)m0 * DDIM);
  tdm_load_16x512_bf16((unsigned)(size_t)&sB[wv][0], kb + (size_t)n0 * DDIM);
  __builtin_amdgcn_s_wait_tensorcnt(0);
  __syncthreads();

  v8f c = {};
  const uint4* qa = (const uint4*)(sQ + l15 * DDIM);
  const uint4* ka = (const uint4*)(&sB[wv][0] + l15 * DDIM);
  for (int kk = 0; kk < DDIM; kk += 32) {
    AFrag a, bm;
    // A (16x32 bf16): lanes 0-15 -> K 0..7 & 16..23 ; lanes 16-31 -> +8
    a.q[0] = qa[(kk + kh * 8) >> 3];
    a.q[1] = qa[(kk + 16 + kh * 8) >> 3];
    // B (32x16 bf16): column = seq_k row n0+l15; lanes 0-15 -> K 0..7 & 16..23 ; lanes 16-31 -> +8
    bm.q[0] = ka[(kk + kh * 8) >> 3];
    bm.q[1] = ka[(kk + 16 + kh * 8) >> 3];
    c = __builtin_amdgcn_wmma_f32_16x16x32_bf16(false, a.v, false, bm.v,
                                                (short)0, c, false, false);
  }
  const int n = n0 + l15;
  const float knv = kn[n];
#pragma unroll
  for (int r = 0; r < 8; ++r) {
    int m = m0 + r + 8 * kh;
    float d2 = qn[m] + knv - 2.0f * c[r];
    Dmat[(size_t)m * TK + n] = sqrtf(fmaxf(d2, 0.0f));
  }
}

// ============================================================================
// Kernel 2: fully fused conv1(relu)->conv2(relu)->conv3 + D residual, tiled.
// Output tile 16x32. Implicit GEMM on v_wmma_f32_16x16x32_bf16.
//   conv1: K packs (tap,channel) pairs -> 2 K-steps of 32
//   conv2: K = 32 input channels, accumulated over 25 taps
// ============================================================================
#define IN_H 26
#define IN_W 48
#define X1_H 22
#define X1_W 48
#define X2_H 18
#define X2_W 48

__global__ __launch_bounds__(512)
void conv_fused_kernel(const float* __restrict__ Dmat,
                       const float* __restrict__ lenq, const float* __restrict__ lenk,
                       const float* __restrict__ c1w, const float* __restrict__ c1b,
                       const float* __restrict__ c2w, const float* __restrict__ c2b,
                       const float* __restrict__ c3w, const float* __restrict__ c3b,
                       float* __restrict__ Sout) {
  __shared__ alignas(16) unsigned       sIn[IN_H * IN_W];       // (bf16 D, bf16 P)
  __shared__ alignas(16) unsigned short sX1[X1_H * X1_W * 32];  // conv1 out
  __shared__ alignas(16) unsigned short sX2[X2_H * X2_W * 32];  // conv2 out
  __shared__ alignas(16) unsigned short sW2[25 * 32 * 32];      // [tap][co][ci]
  __shared__ alignas(16) unsigned short sW1[32 * 64];           // [co][k=tap*2+ch]
  __shared__ alignas(16) float          sW3[9 * 32];            // [tap][ci]
  __shared__ float sB1[32], sB2[32];

  const int tid  = threadIdx.x;
  const int wv   = tid >> 5;
  const int lane = tid & 31;
  const int kh   = lane >> 4;
  const int l15  = lane & 15;
  const int x0 = blockIdx.x * 32;
  const int y0 = blockIdx.y * 16;

  // Prefetch residual D rows for this tile into cache early.
  if (tid < 16) __builtin_prefetch(Dmat + (size_t)(y0 + tid) * TK + x0, 0, 3);

  // ---- stage weights into LDS (padded to 32 channels, co-major for B frags)
  for (int i = tid; i < 32 * 64; i += 512) {
    int co = i >> 6, k = i & 63, t = k >> 1, ch = k & 1;
    float w = (co < 30 && t < 25) ? c1w[co * 50 + ch * 25 + t] : 0.0f;
    sW1[i] = f2bf(w);
  }
  for (int i = tid; i < 25 * 32 * 32; i += 512) {
    int t = i >> 10, co = (i >> 5) & 31, ci = i & 31;
    float w = (co < 30 && ci < 30) ? c2w[(co * 30 + ci) * 25 + t] : 0.0f;
    sW2[i] = f2bf(w);
  }
  for (int i = tid; i < 9 * 32; i += 512) {
    int t = i >> 5, ci = i & 31;
    sW3[i] = (ci < 30) ? c3w[ci * 9 + t] : 0.0f;
  }
  if (tid < 32) {
    sB1[tid] = (tid < 30) ? c1b[tid] : 0.0f;
    sB2[tid] = (tid < 30) ? c2b[tid] : 0.0f;
  }

  // ---- stage input halo: D (bf16) and P=|lq-lk| (bf16), zero-padded
  for (int i = tid; i < IN_H * IN_W; i += 512) {
    int ly = i / IN_W, lx = i - ly * IN_W;
    int gy = y0 - 5 + ly, gx = x0 - 5 + lx;
    float d = 0.0f, p = 0.0f;
    if (gy >= 0 && gy < TQ && gx >= 0 && gx < TK) {
      d = Dmat[(size_t)gy * TK + gx];
      p = fabsf(lenq[gy] - lenk[gx]);
    }
    sIn[i] = (unsigned)f2bf(d) | ((unsigned)f2bf(p) << 16);
  }
  __syncthreads();

  // ---- conv1: 22 rows x 3 groups (xbase 0,16,22) -> X1[22][0..37][32]
  for (int g = wv; g < 66; g += 16) {
    int ly = g / 3, gm = g - ly * 3;
    int xb = (gm == 0) ? 0 : (gm == 1) ? 16 : 22;
    int px = xb + l15;                // pixel of this lane's A row (M)
    v8f c0 = {}, c1 = {};
#pragma unroll
    for (int s = 0; s < 2; ++s) {
      AFrag a;
#pragma unroll
      for (int v = 0; v < 8; ++v) {
        int t  = s * 16 + kh * 8 + v;  // tap of this K pair
        int tc = (t < 25) ? t : 0;
        int dy = tc / 5, dx = tc - dy * 5;
        unsigned val = sIn[(ly + dy) * IN_W + px + dx];     // (D,P) of one tap
        a.u[v] = (t < 25) ? val : 0u;
      }
      AFrag b0, b1;
      const uint4* p0 = (const uint4*)(sW1 + l15 * 64 + s * 32 + kh * 16);
      const uint4* p1 = (const uint4*)(sW1 + (16 + l15) * 64 + s * 32 + kh * 16);
      b0.q[0] = p0[0]; b0.q[1] = p0[1];
      b1.q[0] = p1[0]; b1.q[1] = p1[1];
      c0 = __builtin_amdgcn_wmma_f32_16x16x32_bf16(false, a.v, false, b0.v,
                                                   (short)0, c0, false, false);
      c1 = __builtin_amdgcn_wmma_f32_16x16x32_bf16(false, a.v, false, b1.v,
                                                   (short)0, c1, false, false);
    }
    float bia0 = sB1[l15], bia1 = sB1[16 + l15];
#pragma unroll
    for (int r = 0; r < 8; ++r) {
      int x = xb + r + 8 * kh;
      int base = (ly * X1_W + x) * 32;
      sX1[base + l15]      = f2bf(fmaxf(c0[r] + bia0, 0.0f));
      sX1[base + 16 + l15] = f2bf(fmaxf(c1[r] + bia1, 0.0f));
    }
  }
  __syncthreads();

  // ---- conv2: 18 rows x 3 groups (xbase 0,16,18), 25 taps x K=32 channels
  for (int g = wv; g < 54; g += 16) {
    int r2 = g / 3, gm = g - r2 * 3;
    int xb = (gm == 0) ? 0 : (gm == 1) ? 16 : 18;
    v8f c0 = {}, c1 = {};
    for (int t = 0; t < 25; ++t) {
      int dy = t / 5, dx = t - dy * 5;
      AFrag a, b0, b1;
      const uint4* pa = (const uint4*)(sX1 + ((r2 + dy) * X1_W + xb + l15 + dx) * 32);
      a.q[0] = pa[kh * 2];        // ci 0..7  (+16 for upper half)
      a.q[1] = pa[kh * 2 + 1];    // ci 8..15 (+16 for upper half)
      const uint4* p0 = (const uint4*)(sW2 + (t * 32 + l15) * 32 + kh * 16);
      const uint4* p1 = (const uint4*)(sW2 + (t * 32 + 16 + l15) * 32 + kh * 16);
      b0.q[0] = p0[0]; b0.q[1] = p0[1];
      b1.q[0] = p1[0]; b1.q[1] = p1[1];
      c0 = __builtin_amdgcn_wmma_f32_16x16x32_bf16(false, a.v, false, b0.v,
                                                   (short)0, c0, false, false);
      c1 = __builtin_amdgcn_wmma_f32_16x16x32_bf16(false, a.v, false, b1.v,
                                                   (short)0, c1, false, false);
    }
    float bia0 = sB2[l15], bia1 = sB2[16 + l15];
#pragma unroll
    for (int r = 0; r < 8; ++r) {
      int x = xb + r + 8 * kh;
      int base = (r2 * X2_W + x) * 32;
      sX2[base + l15]      = f2bf(fmaxf(c0[r] + bia0, 0.0f));
      sX2[base + 16 + l15] = f2bf(fmaxf(c1[r] + bia1, 0.0f));
    }
  }
  __syncthreads();

  // ---- conv3 (3x3, 30->1) + D residual, one output pixel per thread
  const int oy = tid >> 5, ox = tid & 31;
  float accv = c3b[0];
#pragma unroll
  for (int dy = 0; dy < 3; ++dy)
#pragma unroll
    for (int dx = 0; dx < 3; ++dx) {
      const unsigned* px2 = (const unsigned*)(sX2 + ((oy + dy) * X2_W + ox + dx) * 32);
      const float* w = sW3 + (dy * 3 + dx) * 32;
#pragma unroll
      for (int ci = 0; ci < 15; ++ci) {          // 30 channels as 15 bf16 pairs
        unsigned u = px2[ci];
        accv += bf2f((unsigned short)(u & 0xffffu)) * w[2 * ci]
              + bf2f((unsigned short)(u >> 16))     * w[2 * ci + 1];
      }
    }
  const int gy = y0 + oy, gx = x0 + ox;
  Sout[(size_t)gy * TK + gx] = accv + Dmat[(size_t)gy * TK + gx];
}

// ============================================================================
// Kernel 3: row softmax of -S (in place in d_out) + fused per-row D.a sums.
// ============================================================================
__global__ __launch_bounds__(256)
void softmax_kernel(float* __restrict__ A, const float* __restrict__ Dmat,
                    float* __restrict__ acc) {
  const int row = blockIdx.x, tid = threadIdx.x;
  float* Sp = A + (size_t)row * TK;
  const float* Dp = Dmat + (size_t)row * TK;
  __shared__ float red[256];

  float m = -1e30f;
  for (int j = tid; j < TK; j += 256) m = fmaxf(m, -Sp[j]);
  red[tid] = m; __syncthreads();
  for (int s = 128; s > 0; s >>= 1) {
    if (tid < s) red[tid] = fmaxf(red[tid], red[tid + s]);
    __syncthreads();
  }
  m = red[0]; __syncthreads();

  float se = 0.0f, sde = 0.0f;
  for (int j = tid; j < TK; j += 256) {
    float e = __expf(-Sp[j] - m);
    Sp[j] = e;
    se  += e;
    sde += Dp[j] * e;
  }
  red[tid] = se; __syncthreads();
  for (int s = 128; s > 0; s >>= 1) {
    if (tid < s) red[tid] += red[tid + s];
    __syncthreads();
  }
  se = red[0]; __syncthreads();
  red[tid] = sde; __syncthreads();
  for (int s = 128; s > 0; s >>= 1) {
    if (tid < s) red[tid] += red[tid + s];
    __syncthreads();
  }
  sde = red[0];

  const float inv = 1.0f / se;
  for (int j = tid; j < TK; j += 256) Sp[j] *= inv;
  if (tid == 0) {
    atomicAdd(&acc[0], sde * inv);   // sum_j D*a for this row
    atomicAdd(&acc[1], 1.0f);        // L1 mass of this row (softmax sums to 1)
  }
}

__global__ void finalize_kernel(const float* __restrict__ acc, float* __restrict__ out) {
  out[0] = acc[0] / acc[1];
}

// ============================================================================
extern "C" void kernel_launch(void* const* d_in, const int* in_sizes, int n_in,
                              void* d_out, int out_size, void* d_ws, size_t ws_size,
                              hipStream_t stream) {
  const float* seq_q = (const float*)d_in[0];
  const float* seq_k = (const float*)d_in[1];
  const float* len_q = (const float*)d_in[2];
  const float* len_k = (const float*)d_in[3];
  const float* c1w   = (const float*)d_in[4];
  const float* c1b   = (const float*)d_in[5];
  const float* c2w   = (const float*)d_in[6];
  const float* c2b   = (const float*)d_in[7];
  const float* c3w   = (const float*)d_in[8];
  const float* c3b   = (const float*)d_in[9];

  char* ws = (char*)d_ws;
  unsigned short* qb = (unsigned short*)ws;                              // 2 MB
  unsigned short* kb = (unsigned short*)(ws + (2ull << 20));             // 2 MB
  float* qn   = (float*)(ws + (4ull << 20));                             // 8 KB
  float* kn   = (float*)(ws + (4ull << 20) + 8192);                      // 8 KB
  float* Dmat = (float*)(ws + (4ull << 20) + 16384);                     // 16 MB
  float* acc  = (float*)(ws + (4ull << 20) + 16384 + (16ull << 20));     // 8 B

  float* A   = (float*)d_out;                   // [2048*2048] attention matrix
  float* dis = A + (size_t)TQ * TK;             // scalar

  prep_kernel<<<TQ + TK + 1, 256, 0, stream>>>(seq_q, seq_k, qb, kb, qn, kn, acc);
  dist_kernel<<<2048, 256, 0, stream>>>(qb, kb, qn, kn, Dmat);
  dim3 grid2(64, 128);   // 32-wide x 16-tall output tiles over 2048x2048
  conv_fused_kernel<<<grid2, 512, 0, stream>>>(Dmat, len_q, len_k,
                                               c1w, c1b, c2w, c2b, c3w, c3b, A);
  softmax_kernel<<<TQ, 256, 0, stream>>>(A, Dmat, acc);
  finalize_kernel<<<1, 1, 0, stream>>>(acc, dis);
}
